// GAT_51187420233863
// MI455X (gfx1250) — matrix operand order repaired and checked by reading, not compiled, not run on hardware
//
#include <hip/hip_runtime.h>
#include <stdint.h>

// ---------------------------------------------------------------------------
// GATv2 x3 on gfx1250 (MI455X).
//  - Dense node transforms: dual-B bf16 WMMA GEMM (v_wmma_f32_16x16x32_bf16),
//    f32 accumulation, vectorized fp32->bf16 LDS staging, A-tile prefetch.
//  - Edge attention: 3-pass segment softmax with atomics (self-loops appended).
// ---------------------------------------------------------------------------

typedef __attribute__((ext_vector_type(16))) __bf16 bf16x16;
typedef __attribute__((ext_vector_type(8)))  float  f32x8;

#define TM 128
#define TN 64
#define TK 32
#define LDT 40   // padded LDS row stride (bf16 elems): 80B = 16B-aligned, conflict-free

// =============================== WMMA GEMM =================================
// C0 = A*B0, C1 = A*B1 ; A[M x K], B[K x Nmat] fp32 in global -> bf16 tiles.
__global__ __launch_bounds__(256)
void gemm_bf16_wmma_dual(const float* __restrict__ A,
                         const float* __restrict__ B0g, const float* __restrict__ B1g,
                         float* __restrict__ C0, float* __restrict__ C1,
                         int M, int Nmat, int K) {
  __shared__ __align__(16) __bf16 sA [TM * LDT];
  __shared__ __align__(16) __bf16 sB0[TN * LDT];
  __shared__ __align__(16) __bf16 sB1[TN * LDT];
  const int t    = threadIdx.x;
  const int lane = t & 31;
  const int wave = t >> 5;
  const int wm   = (wave & 3) * 32;   // wave row offset within block tile
  const int wn   = (wave >> 2) * 32;  // wave col offset within block tile
  const int rowBase = blockIdx.x * TM;
  const int colBase = blockIdx.y * TN;
  const bool fullN = ((Nmat & 63) == 0) && (colBase + TN <= Nmat);

  // ---- loop-invariant staging coordinates ----
  const int  am    = t >> 1;            // A tile row 0..127
  const int  akk   = (t & 1) * 16;      // A tile k half: 0 or 16
  const bool aRowOk = (rowBase + am) < M;
  const float* aPtr = A + (size_t)(rowBase + am) * K + akk;  // + k0 each iter

  const int bn0 = (t & 15) * 4;         // B col group (4 cols)
  const int bk  = t >> 4;               // B k row 0..15 (also handles +16)

  f32x8 c0_00 = {}, c0_01 = {}, c0_10 = {}, c0_11 = {};
  f32x8 c1_00 = {}, c1_01 = {}, c1_10 = {}, c1_11 = {};

  for (int k0 = 0; k0 < K; k0 += TK) {
    // --- stage A tile: 16 contiguous floats/thread -> 2x ds_store_b128 ---
    union { __bf16 h[16]; uint4 q[2]; } ar;
    if (aRowOk) {
      float f[16];
#pragma unroll
      for (int j = 0; j < 4; ++j)
        *(float4*)&f[j * 4] = *(const float4*)(aPtr + k0 + j * 4);
#pragma unroll
      for (int j = 0; j < 16; ++j) ar.h[j] = (__bf16)f[j];
    } else {
      ar.q[0] = uint4{0u, 0u, 0u, 0u};
      ar.q[1] = uint4{0u, 0u, 0u, 0u};
    }
    *(uint4*)&sA[am * LDT + akk]     = ar.q[0];
    *(uint4*)&sA[am * LDT + akk + 8] = ar.q[1];

    // --- stage B tiles transposed: sB[n][k] = B[(k0+k)*Nmat + colBase+n] ---
    if (fullN) {
#pragma unroll
      for (int kk = 0; kk < 2; ++kk) {
        int k = bk + kk * 16;
        size_t off = (size_t)(k0 + k) * Nmat + colBase + bn0;
        float4 f0 = *(const float4*)(B0g + off);
        float4 f1 = *(const float4*)(B1g + off);
#pragma unroll
        for (int j = 0; j < 4; ++j) {
          sB0[(bn0 + j) * LDT + k] = (__bf16)((const float*)&f0)[j];
          sB1[(bn0 + j) * LDT + k] = (__bf16)((const float*)&f1)[j];
        }
      }
    } else {
#pragma unroll
      for (int i = 0; i < (TN * TK) / 256; ++i) {
        int idx = t + i * 256;
        int n = idx & 63, k = idx >> 6;
        int gc = colBase + n;
        float v0 = 0.f, v1 = 0.f;
        if (gc < Nmat) {
          size_t off = (size_t)(k0 + k) * Nmat + gc;
          v0 = B0g[off];
          v1 = B1g[off];
        }
        sB0[n * LDT + k] = (__bf16)v0;
        sB1[n * LDT + k] = (__bf16)v1;
      }
    }
    __syncthreads();

    // prefetch next A tile into cache while WMMAs run (global_prefetch_b8)
    if (aRowOk && (k0 + TK < K)) __builtin_prefetch(aPtr + k0 + TK, 0, 3);

    // --- build fragments in contiguous k order (k = 16h + s), same for A and B ---
    // A 16x32 bf16: lanes0-15 row=base+lane K{0..15};       lanes16-31 K{16..31}
    // B 32x16 bf16: lanes0-15 col=base+lane K{0..15};       lanes16-31 K{16..31}
    union Frag { bf16x16 v; uint4 q[2]; };
    Frag a0, a1, p0, p1, q0, q1;
    {
      int lrow = lane & 15;
      int ka = (lane < 16) ? 0 : 16;
      const __bf16* pa0 = &sA[(wm + lrow) * LDT + ka];
      const __bf16* pa1 = &sA[(wm + 16 + lrow) * LDT + ka];
      a0.q[0] = *(const uint4*)(pa0);       a0.q[1] = *(const uint4*)(pa0 + 8);
      a1.q[0] = *(const uint4*)(pa1);       a1.q[1] = *(const uint4*)(pa1 + 8);
      int kb = (lane < 16) ? 0 : 16;
      const __bf16* pb0 = &sB0[(wn + lrow) * LDT + kb];
      const __bf16* pb1 = &sB0[(wn + 16 + lrow) * LDT + kb];
      p0.q[0] = *(const uint4*)(pb0);       p0.q[1] = *(const uint4*)(pb0 + 8);
      p1.q[0] = *(const uint4*)(pb1);       p1.q[1] = *(const uint4*)(pb1 + 8);
      const __bf16* pc0 = &sB1[(wn + lrow) * LDT + kb];
      const __bf16* pc1 = &sB1[(wn + 16 + lrow) * LDT + kb];
      q0.q[0] = *(const uint4*)(pc0);       q0.q[1] = *(const uint4*)(pc0 + 8);
      q1.q[0] = *(const uint4*)(pc1);       q1.q[1] = *(const uint4*)(pc1 + 8);
    }
    c0_00 = __builtin_amdgcn_wmma_f32_16x16x32_bf16(false, a0.v, false, p0.v, (short)0, c0_00, false, false);
    c0_01 = __builtin_amdgcn_wmma_f32_16x16x32_bf16(false, a0.v, false, p1.v, (short)0, c0_01, false, false);
    c0_10 = __builtin_amdgcn_wmma_f32_16x16x32_bf16(false, a1.v, false, p0.v, (short)0, c0_10, false, false);
    c0_11 = __builtin_amdgcn_wmma_f32_16x16x32_bf16(false, a1.v, false, p1.v, (short)0, c0_11, false, false);
    c1_00 = __builtin_amdgcn_wmma_f32_16x16x32_bf16(false, a0.v, false, q0.v, (short)0, c1_00, false, false);
    c1_01 = __builtin_amdgcn_wmma_f32_16x16x32_bf16(false, a0.v, false, q1.v, (short)0, c1_01, false, false);
    c1_10 = __builtin_amdgcn_wmma_f32_16x16x32_bf16(false, a1.v, false, q0.v, (short)0, c1_10, false, false);
    c1_11 = __builtin_amdgcn_wmma_f32_16x16x32_bf16(false, a1.v, false, q1.v, (short)0, c1_11, false, false);
    __syncthreads();
  }

  // C layout: VGPR r -> lanes0-15 (M=r, N=lane); lanes16-31 (M=r+8, N=lane-16)
  int ln = lane & 15;
  int mAdd = (lane < 16) ? 0 : 8;
#pragma unroll
  for (int r = 0; r < 8; ++r) {
    int row0 = rowBase + wm + r + mAdd;
    int row1 = rowBase + wm + 16 + r + mAdd;
    int col0 = colBase + wn + ln;
    int col1 = colBase + wn + 16 + ln;
    if (row0 < M) {
      if (col0 < Nmat) { C0[(size_t)row0 * Nmat + col0] = c0_00[r]; C1[(size_t)row0 * Nmat + col0] = c1_00[r]; }
      if (col1 < Nmat) { C0[(size_t)row0 * Nmat + col1] = c0_01[r]; C1[(size_t)row0 * Nmat + col1] = c1_01[r]; }
    }
    if (row1 < M) {
      if (col0 < Nmat) { C0[(size_t)row1 * Nmat + col0] = c0_10[r]; C1[(size_t)row1 * Nmat + col0] = c1_10[r]; }
      if (col1 < Nmat) { C0[(size_t)row1 * Nmat + col1] = c0_11[r]; C1[(size_t)row1 * Nmat + col1] = c1_11[r]; }
    }
  }
}

// ============================ edge-phase kernels ===========================
// order-preserving float<->uint map so atomicMax(unsigned) == float max
__device__ __forceinline__ unsigned f2mono(float f) {
  unsigned u = __float_as_uint(f);
  return (u & 0x80000000u) ? ~u : (u | 0x80000000u);
}
__device__ __forceinline__ float mono2f(unsigned e) {
  return (e & 0x80000000u) ? __uint_as_float(e & 0x7FFFFFFFu)
                           : __uint_as_float(~e);
}

// pass 1: logits[e,h] = att[h].LeakyReLU(xl[src]+xr[dst]); segment-max into mbuf
__global__ __launch_bounds__(256)
void edge_logits_kernel(const float* __restrict__ xl, const float* __restrict__ xr,
                        const float* __restrict__ att, const long long* __restrict__ ei,
                        int E, int Nn, int H, int F,
                        float* __restrict__ logits, unsigned* __restrict__ mbuf) {
  long long gid = (((long long)blockIdx.x * blockDim.x) + threadIdx.x) >> 5;
  int lane = threadIdx.x & 31;
  long long E2 = (long long)E + Nn;
  if (gid >= E2 * H) return;
  int h = (int)(gid % H);
  long long e = gid / H;
  long long s = (e < E) ? ei[e]     : (e - E);
  long long d = (e < E) ? ei[E + e] : (e - E);
  int HF = H * F;
  const float* pl = xl + s * HF + (size_t)h * F;
  const float* pr = xr + d * HF + (size_t)h * F;
  const float* pa = att + (size_t)h * F;
  float acc = 0.f;
  for (int f = lane; f < F; f += 32) {
    float v = pl[f] + pr[f];
    v = (v > 0.f) ? v : 0.2f * v;       // LeakyReLU slope 0.2
    acc += pa[f] * v;
  }
  for (int off = 16; off > 0; off >>= 1) acc += __shfl_xor(acc, off, 32);
  if (lane == 0) {
    logits[e * H + h] = acc;
    atomicMax(&mbuf[d * H + h], f2mono(acc));
  }
}

// pass 2: in-place exp(logit - seg_max), accumulate segment denominators
__global__ __launch_bounds__(256)
void edge_expsum_kernel(float* __restrict__ logits, const unsigned* __restrict__ mbuf,
                        float* __restrict__ dbuf, const long long* __restrict__ ei,
                        int E, int Nn, int H) {
  long long idx = (long long)blockIdx.x * blockDim.x + threadIdx.x;
  long long E2 = (long long)E + Nn;
  if (idx >= E2 * H) return;
  int h = (int)(idx % H);
  long long e = idx / H;
  long long d = (e < E) ? ei[E + e] : (e - E);
  float m = mono2f(mbuf[d * H + h]);
  float ex = __expf(logits[idx] - m);
  logits[idx] = ex;
  atomicAdd(&dbuf[d * H + h], ex);
}

// pass 3: out[dst] += (alpha/denom) * xl[src]
__global__ __launch_bounds__(256)
void edge_aggregate_kernel(const float* __restrict__ xl, const float* __restrict__ alpha,
                           const float* __restrict__ dbuf, const long long* __restrict__ ei,
                           int E, int Nn, int H, int F, float* __restrict__ out) {
  long long gid = (((long long)blockIdx.x * blockDim.x) + threadIdx.x) >> 5;
  int lane = threadIdx.x & 31;
  long long E2 = (long long)E + Nn;
  if (gid >= E2 * H) return;
  int h = (int)(gid % H);
  long long e = gid / H;
  long long s = (e < E) ? ei[e]     : (e - E);
  long long d = (e < E) ? ei[E + e] : (e - E);
  int HF = H * F;
  float w = alpha[e * H + h] / dbuf[d * H + h];
  const float* pl = xl + s * HF + (size_t)h * F;
  float* po = out + d * HF + (size_t)h * F;
  for (int f = lane; f < F; f += 32) atomicAdd(&po[f], w * pl[f]);
}

__global__ __launch_bounds__(256)
void relu_bias_kernel(const float* __restrict__ in, const float* __restrict__ b,
                      float* __restrict__ out, long long total, int C) {
  long long i = (long long)blockIdx.x * blockDim.x + threadIdx.x;
  if (i >= total) return;
  float v = in[i] + b[i % C];
  out[i] = v > 0.f ? v : 0.f;
}

__global__ __launch_bounds__(256)
void bias_add_kernel(float* __restrict__ out, const float* __restrict__ b,
                     long long total, int C) {
  long long i = (long long)blockIdx.x * blockDim.x + threadIdx.x;
  if (i >= total) return;
  out[i] += b[i % C];
}

// ================================ launcher =================================
static inline long long cdivll(long long a, long long b) { return (a + b - 1) / b; }

extern "C" void kernel_launch(void* const* d_in, const int* in_sizes, int n_in,
                              void* d_out, int out_size, void* d_ws, size_t ws_size,
                              hipStream_t stream) {
  const float*     x    = (const float*)d_in[0];
  const long long* ei   = (const long long*)d_in[1];  // int64 (2,E)
  const float*     Wl1  = (const float*)d_in[2];
  const float*     Wr1  = (const float*)d_in[3];
  const float*     att1 = (const float*)d_in[4];
  const float*     b1   = (const float*)d_in[5];
  const float*     Wl2  = (const float*)d_in[6];
  const float*     Wr2  = (const float*)d_in[7];
  const float*     att2 = (const float*)d_in[8];
  const float*     b2   = (const float*)d_in[9];
  const float*     Wl3  = (const float*)d_in[10];
  const float*     Wr3  = (const float*)d_in[11];
  const float*     att3 = (const float*)d_in[12];
  const float*     b3   = (const float*)d_in[13];

  const int DIN = 512, HF1 = 512, H1n = 8, F1 = 64, NOUT2 = 40, NOUT3 = 2;
  const int Nn = in_sizes[0] / DIN;          // 50000
  const int E  = in_sizes[1] / 2;            // 400000
  const long long E2 = (long long)E + Nn;    // + self loops

  // ---- workspace layout (floats) ----
  float* ws = (float*)d_ws;
  const size_t NS = (size_t)Nn * HF1;
  float* XL1     = ws;                 // later reused for layer-2/3 transforms
  float* XR1     = ws + NS;            // reused as OUT1 after its last read
  float* OUT1    = XR1;
  float* Hh      = ws + 2 * NS;        // relu(gat1+b1)
  float* logits1 = ws + 3 * NS;                            // E2*8
  float* STAT    = logits1 + (size_t)E2 * H1n;
  unsigned* m1   = (unsigned*)STAT;                        // N*8
  float*    d1   = STAT + (size_t)Nn * 8;                  // N*8
  unsigned* m2   = (unsigned*)(STAT + (size_t)Nn * 16);    // N
  float*    d2   = STAT + (size_t)Nn * 17;                 // N
  unsigned* m3   = (unsigned*)(STAT + (size_t)Nn * 18);    // N
  float*    d3   = STAT + (size_t)Nn * 19;                 // N
  float* logits2 = STAT + (size_t)Nn * 20;                 // E2
  float* logits3 = logits2 + E2;                           // E2
  // layer-2/3 transforms recycle the XL1 region:
  float* XL2 = ws;
  float* XR2 = ws + (size_t)Nn * NOUT2;
  float* XL3 = ws + (size_t)Nn * NOUT2 * 2;
  float* XR3 = XL3 + (size_t)Nn * NOUT3;

  float* yClass  = (float*)d_out;                       // N x 40
  float* yDomain = (float*)d_out + (size_t)Nn * NOUT2;  // N x 2

  // ---- zero init (monotone-encoded 0 == -inf sentinel for the max bufs) ----
  hipMemsetAsync(STAT, 0, (size_t)Nn * 20 * sizeof(float), stream);
  hipMemsetAsync(d_out, 0, (size_t)out_size * sizeof(float), stream);

  dim3 blk(256);

  // ---- layer 1: node transforms (dual-B bf16 WMMA) ----
  dim3 g1((unsigned)cdivll(Nn, TM), (unsigned)cdivll(HF1, TN));
  gemm_bf16_wmma_dual<<<g1, blk, 0, stream>>>(x, Wl1, Wr1, XL1, XR1, Nn, HF1, DIN);

  // ---- layer 1: attention ----
  long long w1 = E2 * H1n;  // one wave per (edge, head)
  edge_logits_kernel<<<(unsigned)cdivll(w1 * 32, 256), blk, 0, stream>>>(
      XL1, XR1, att1, ei, E, Nn, H1n, F1, logits1, m1);
  edge_expsum_kernel<<<(unsigned)cdivll(w1, 256), blk, 0, stream>>>(
      logits1, m1, d1, ei, E, Nn, H1n);
  hipMemsetAsync(OUT1, 0, NS * sizeof(float), stream);  // XR1 dead after logits pass
  edge_aggregate_kernel<<<(unsigned)cdivll(w1 * 32, 256), blk, 0, stream>>>(
      XL1, logits1, d1, ei, E, Nn, H1n, F1, OUT1);
  relu_bias_kernel<<<(unsigned)cdivll((long long)NS, 256), blk, 0, stream>>>(
      OUT1, b1, Hh, (long long)NS, HF1);

  // ---- layer 2/3: node transforms from Hh (XL1 region now recyclable) ----
  dim3 g2((unsigned)cdivll(Nn, TM), 1);
  gemm_bf16_wmma_dual<<<g2, blk, 0, stream>>>(Hh, Wl2, Wr2, XL2, XR2, Nn, NOUT2, HF1);
  gemm_bf16_wmma_dual<<<g2, blk, 0, stream>>>(Hh, Wl3, Wr3, XL3, XR3, Nn, NOUT3, HF1);

  // ---- layer 2 attention -> y_class ----
  edge_logits_kernel<<<(unsigned)cdivll(E2 * 32, 256), blk, 0, stream>>>(
      XL2, XR2, att2, ei, E, Nn, 1, NOUT2, logits2, m2);
  edge_expsum_kernel<<<(unsigned)cdivll(E2, 256), blk, 0, stream>>>(
      logits2, m2, d2, ei, E, Nn, 1);
  edge_aggregate_kernel<<<(unsigned)cdivll(E2 * 32, 256), blk, 0, stream>>>(
      XL2, logits2, d2, ei, E, Nn, 1, NOUT2, yClass);
  bias_add_kernel<<<(unsigned)cdivll((long long)Nn * NOUT2, 256), blk, 0, stream>>>(
      yClass, b2, (long long)Nn * NOUT2, NOUT2);

  // ---- layer 3 attention -> y_domain ----
  edge_logits_kernel<<<(unsigned)cdivll(E2 * 32, 256), blk, 0, stream>>>(
      XL3, XR3, att3, ei, E, Nn, 1, NOUT3, logits3, m3);
  edge_expsum_kernel<<<(unsigned)cdivll(E2, 256), blk, 0, stream>>>(
      logits3, m3, d3, ei, E, Nn, 1);
  edge_aggregate_kernel<<<(unsigned)cdivll(E2 * 32, 256), blk, 0, stream>>>(
      XL3, logits3, d3, ei, E, Nn, 1, NOUT3, yDomain);
  bias_add_kernel<<<(unsigned)cdivll((long long)Nn * NOUT3, 256), blk, 0, stream>>>(
      yDomain, b3, (long long)Nn * NOUT3, NOUT3);
}
